// GCN_54726473286012
// MI455X (gfx1250) — matrix operand order repaired and checked by reading, not compiled, not run on hardware
//
#include <hip/hip_runtime.h>
#include <hip/hip_bf16.h>

// ---------- types ----------
typedef __bf16 bf16;
typedef bf16  v16bf __attribute__((ext_vector_type(16)));
typedef float v8f   __attribute__((ext_vector_type(8)));

union ABfrag { v16bf v; uint4 q[2]; };         // A fragment = 2 x 16-byte raw loads

// pack two f32 -> two bf16 (RNE) in one dword: lo in [15:0], hi in [31:16].
__device__ __forceinline__ unsigned pack2bf(float lo, float hi) {
    union { float f; unsigned u; } a, b; a.f = lo; b.f = hi;
    unsigned ua = a.u + (0x7fffu + ((a.u >> 16) & 1u));
    unsigned ub = b.u + (0x7fffu + ((b.u >> 16) & 1u));
    return __builtin_amdgcn_perm(ub, ua, 0x07060302u);
}

__device__ __forceinline__ bf16 f2bf(float f) {
    union { float f; unsigned u; } in; in.f = f;
    unsigned u = in.u + (0x7fffu + ((in.u >> 16) & 1u));
    union { unsigned short s; bf16 b; } out;
    out.s = (unsigned short)(u >> 16);
    return out.b;
}

// ---------- degree / normalization ----------
__global__ void k_ones(float* deg, int n) {
    int i = blockIdx.x * blockDim.x + threadIdx.x;
    if (i < n) deg[i] = 1.0f;                   // self loop
}

__global__ void k_count(const long long* __restrict__ rows, float* deg, int ne) {
    int e = blockIdx.x * blockDim.x + threadIdx.x;
    if (e < ne) {
        int r = (int)rows[e];
        __hip_atomic_fetch_add(deg + r, 1.0f, __ATOMIC_RELAXED, __HIP_MEMORY_SCOPE_AGENT);
    }
}

__global__ void k_rsqrt(float* d, int n) {
    int i = blockIdx.x * blockDim.x + threadIdx.x;
    if (i < n) d[i] = rsqrtf(d[i]);             // deg >= 1 always
}

// ---------- streaming f32 -> bf16 (optional relu), float4 -> 4x bf16 per thread ----------
template<bool RELU>
__global__ void k_cvt_bf16(const float* __restrict__ src, bf16* __restrict__ dst, long long n4) {
    long long i = (long long)blockIdx.x * blockDim.x + threadIdx.x;
    if (i >= n4) return;
    float4 v = ((const float4*)src)[i];
    if (RELU) {
        v.x = fmaxf(v.x, 0.0f); v.y = fmaxf(v.y, 0.0f);
        v.z = fmaxf(v.z, 0.0f); v.w = fmaxf(v.w, 0.0f);
    }
    uint2 o;
    o.x = pack2bf(v.x, v.y);
    o.y = pack2bf(v.z, v.w);
    ((uint2*)dst)[i] = o;
}

// ---------- dense GEMM: T[N,NCOL] = Xbf[N,128] @ W[128,NCOL] via bf16 WMMA ----------
// A is already bf16 row-major: fragments are raw 16-byte loads, zero conversion VALU.
// 256 threads = 8 wave32; each wave computes 16 rows x NCOL cols.
template<int NCOL>
__global__ __launch_bounds__(256)
void k_gemm_bf16(const bf16* __restrict__ X, const float* __restrict__ W,
                 float* __restrict__ T, int nrows) {
    constexpr int NT = NCOL / 16;               // 16-col output tiles
    __shared__ bf16 Wt[NCOL * 128];             // W transposed: Wt[n*128 + k]

    const int tid = threadIdx.x;
    for (int i = tid; i < NCOL * 128; i += 256) {
        int n = i >> 7, k = i & 127;
        Wt[i] = f2bf(W[k * NCOL + n]);
    }
    __syncthreads();

    const int wave = tid >> 5;
    const int lane = tid & 31;
    const int half = lane >> 4;                 // 0: lanes 0-15, 1: lanes 16-31
    const int lrow = lane & 15;
    const int rowBase = blockIdx.x * 128 + wave * 16;
    const int row  = rowBase + lrow;
    // clamp OOB lanes to a valid row: their A row only feeds D rows we never store
    const int rowc = row < nrows ? row : (nrows - 1);
    const uint4* __restrict__ xv = (const uint4*)(X + (size_t)rowc * 128);  // 16 uint4 / row

    // A fragment layout (16-bit A 16x32), K-block kk = 32*s:
    //   elements 0..7  -> K = kk + half*16 + e       (one aligned 16B load)
    //   elements 8..15 -> K = kk + half*16 + e       (one aligned 16B load)
    ABfrag a[4];
#pragma unroll
    for (int s = 0; s < 4; ++s) {
        const int k0 = 32 * s + half * 16;
        a[s].q[0] = xv[k0 >> 3];
        a[s].q[1] = xv[(k0 + 8) >> 3];
    }

    v8f acc[NT];
#pragma unroll
    for (int t = 0; t < NT; ++t)
#pragma unroll
        for (int r = 0; r < 8; ++r) acc[t][r] = 0.0f;

#pragma unroll
    for (int s = 0; s < 4; ++s) {
        const int kk = 32 * s;
#pragma unroll
        for (int t = 0; t < NT; ++t) {
            // B fragment, 32x16 layout: n = lrow, element e -> K = kk + half*16 + e
            const v16bf b = *(const v16bf*)&Wt[(t * 16 + lrow) * 128 + kk + half * 16];
            acc[t] = __builtin_amdgcn_wmma_f32_16x16x32_bf16(
                false, a[s].v, false, b, (short)0, acc[t], false, false);
        }
    }

    // C/D layout: VGPR r, lanes 0-15: M=r, lanes 16-31: M=r+8; N=lane%16
#pragma unroll
    for (int t = 0; t < NT; ++t) {
#pragma unroll
        for (int r = 0; r < 8; ++r) {
            int orow = rowBase + half * 8 + r;
            if (orow < nrows)
                T[(size_t)orow * NCOL + t * 16 + lrow] = acc[t][r];
        }
    }
}

// ---------- O[i,f] = bias[f] + dinv[i]^2 * T[i,f]  (bias + self-loop term), float4 ----------
template<int NC>
__global__ void k_init(const float* __restrict__ T, const float* __restrict__ dinv,
                       const float* __restrict__ bias, float* __restrict__ O, int n) {
    long long i4 = (long long)blockIdx.x * blockDim.x + threadIdx.x;   // float4 index
    if (i4 >= (long long)n * (NC / 4)) return;
    int r  = (int)(i4 / (NC / 4));
    int f4 = (int)(i4 % (NC / 4));
    float di = dinv[r];
    float s = di * di;
    float4 t = ((const float4*)T)[i4];
    float4 b = ((const float4*)bias)[f4];
    float4 o;
    o.x = b.x + s * t.x; o.y = b.y + s * t.y;
    o.z = b.z + s * t.z; o.w = b.w + s * t.w;
    ((float4*)O)[i4] = o;
}

// ---------- edge scatter: O[row] += dinv[row]*dinv[col] * T[col], 1 wave / edge ----------
template<int NC>
__global__ __launch_bounds__(256)
void k_scatter(const long long* __restrict__ rows, const long long* __restrict__ cols,
               const float* __restrict__ dinv, const float* __restrict__ T,
               float* __restrict__ O, int ne) {
    const int wave = threadIdx.x >> 5;
    const int lane = threadIdx.x & 31;
    const int e = blockIdx.x * 8 + wave;
    if (e >= ne) return;
    const int r = (int)rows[e];
    const int c = (int)cols[e];
    const float nrm = dinv[r] * dinv[c];
    const float4* __restrict__ src = (const float4*)(T + (size_t)c * NC);
    float* dst = O + (size_t)r * NC;
#pragma unroll
    for (int j = lane; j < NC / 4; j += 32) {
        float4 v = src[j];
        __hip_atomic_fetch_add(dst + 4 * j + 0, nrm * v.x, __ATOMIC_RELAXED, __HIP_MEMORY_SCOPE_AGENT);
        __hip_atomic_fetch_add(dst + 4 * j + 1, nrm * v.y, __ATOMIC_RELAXED, __HIP_MEMORY_SCOPE_AGENT);
        __hip_atomic_fetch_add(dst + 4 * j + 2, nrm * v.z, __ATOMIC_RELAXED, __HIP_MEMORY_SCOPE_AGENT);
        __hip_atomic_fetch_add(dst + 4 * j + 3, nrm * v.w, __ATOMIC_RELAXED, __HIP_MEMORY_SCOPE_AGENT);
    }
}

// ---------- driver ----------
extern "C" void kernel_launch(void* const* d_in, const int* in_sizes, int n_in,
                              void* d_out, int out_size, void* d_ws, size_t ws_size,
                              hipStream_t stream) {
    const float*     x    = (const float*)d_in[0];
    const long long* ei   = (const long long*)d_in[1];
    const float*     W0   = (const float*)d_in[2];
    const float*     b0   = (const float*)d_in[3];
    const float*     W1   = (const float*)d_in[4];
    const float*     b1   = (const float*)d_in[5];
    const float*     W2   = (const float*)d_in[6];
    const float*     b2   = (const float*)d_in[7];

    const int N = in_sizes[0] / 128;            // 100000
    const int E = in_sizes[1] / 2;              // 1600000
    const long long* rows = ei;
    const long long* cols = ei + E;

    // workspace: dinv[N] | T[N*128 f32] | O[N*128 f32] | Hbf[N*128 bf16]  (~128 MB)
    float* dinv = (float*)d_ws;
    float* T    = dinv + (((size_t)N + 63) & ~(size_t)63);
    float* O    = T + (size_t)N * 128;
    bf16*  Hbf  = (bf16*)(O + (size_t)N * 128);

    // normalization: deg (with self loops) -> rsqrt
    k_ones <<<(N + 255) / 256, 256, 0, stream>>>(dinv, N);
    k_count<<<(E + 255) / 256, 256, 0, stream>>>(rows, dinv, E);
    k_rsqrt<<<(N + 255) / 256, 256, 0, stream>>>(dinv, N);

    const int gGemm    = (N + 127) / 128;
    const int gEdge    = (E + 7) / 8;
    const long long n4_128 = (long long)N * 32;                    // N*128/4
    const int gCvt128  = (int)((n4_128 + 255) / 256);
    const int gInit128 = (int)((n4_128 + 255) / 256);
    const int gInit64  = (int)(((long long)N * 16 + 255) / 256);

    // layer 0: Hbf = bf16(x) ; T = Hbf @ W0 ; O = b0 + dinv^2*T + scatter
    k_cvt_bf16<false><<<gCvt128, 256, 0, stream>>>(x, Hbf, n4_128);
    k_gemm_bf16<128><<<gGemm, 256, 0, stream>>>(Hbf, W0, T, N);
    k_init<128><<<gInit128, 256, 0, stream>>>(T, dinv, b0, O, N);
    k_scatter<128><<<gEdge, 256, 0, stream>>>(rows, cols, dinv, T, O, E);

    // layer 1: Hbf = bf16(relu(O)) ; T = Hbf @ W1 ; O = b1 + dinv^2*T + scatter
    k_cvt_bf16<true><<<gCvt128, 256, 0, stream>>>(O, Hbf, n4_128);
    k_gemm_bf16<128><<<gGemm, 256, 0, stream>>>(Hbf, W1, T, N);
    k_init<128><<<gInit128, 256, 0, stream>>>(T, dinv, b1, O, N);
    k_scatter<128><<<gEdge, 256, 0, stream>>>(rows, cols, dinv, T, O, E);

    // layer 2 (no activation on output): Hbf = bf16(relu(O)) ; T = Hbf @ W2 ; d_out
    float* out = (float*)d_out;
    k_cvt_bf16<true><<<gCvt128, 256, 0, stream>>>(O, Hbf, n4_128);
    k_gemm_bf16<64><<<gGemm, 256, 0, stream>>>(Hbf, W2, T, N);
    k_init<64><<<gInit64, 256, 0, stream>>>(T, dinv, b2, out, N);
    k_scatter<64><<<gEdge, 256, 0, stream>>>(rows, cols, dinv, T, out, E);
}
